// DeepSeekMoEModel_33380485824727
// MI455X (gfx1250) — matrix operand order repaired and checked by reading, not compiled, not run on hardware
//
#include <hip/hip_runtime.h>
#include <stdint.h>

// ---------------- problem constants ----------------
#define H     1024
#define IDIM  4096
#define NE    8
#define TOPK  2
#define TKN   4096          // total tokens (2*2048)
#define NSLOT (TKN*TOPK)    // 8192 routed slots
#define BM    128           // block tile M (tokens/slots)
#define BN    64            // block tile N
#define BK    32            // K step (bf16 WMMA K)
#define PK    40            // padded LDS row stride (ushorts) = 80B
                            // bank = row*20 mod 64, gcd(20,64)=4 -> 16 distinct
                            // banks per 16-lane group: conflict-free b128 frags

typedef __attribute__((ext_vector_type(16))) __bf16 bf16x16;
typedef __attribute__((ext_vector_type(8)))  float  floatx8;

union Frag { uint4 q[2]; bf16x16 v; };
static_assert(sizeof(Frag) == 32, "frag size");

// gfx1250 hardware packed fp32->bf16 (RNE): one VALU op for two elements.
__device__ __forceinline__ unsigned int cvt_pk_bf16(float lo, float hi) {
  unsigned int r;
  asm("v_cvt_pk_bf16_f32 %0, %1, %2" : "=v"(r) : "v"(lo), "v"(hi));
  return r;
}
__device__ __forceinline__ unsigned short f2bf(float f) {
  return (unsigned short)(cvt_pk_bf16(f, f) & 0xFFFFu);
}
__device__ __forceinline__ uint4 pack8(float4 f0, float4 f1) {
  uint4 pk;
  pk.x = cvt_pk_bf16(f0.x, f0.y);
  pk.y = cvt_pk_bf16(f0.z, f0.w);
  pk.z = cvt_pk_bf16(f1.x, f1.y);
  pk.w = cvt_pk_bf16(f1.z, f1.w);
  return pk;
}

// CDNA5 async global->LDS copy (GLOBAL_LOAD_ASYNC_TO_LDS_B128, ASYNCcnt-tracked).
// Low 32 bits of the flat address of a __shared__ object are the LDS byte
// address (ISA 10.2).
__device__ __forceinline__ void async_copy_b128(const void* gsrc, void* lds_dst) {
  unsigned int l = (unsigned int)(uintptr_t)lds_dst;
  asm volatile("global_load_async_to_lds_b128 %0, %1, off"
               :: "v"(l), "v"(gsrc) : "memory");
}
__device__ __forceinline__ void wait_async0() {
  asm volatile("s_wait_asynccnt 0" ::: "memory");
}

// ---------------- routing ----------------
__global__ void init_meta(int* counts, int* cursor) {
  int i = threadIdx.x;
  if (i < 16) { counts[i] = 0; cursor[i] = 0; }
}

__global__ void router_kernel(const float* __restrict__ x, const float* __restrict__ rw,
                              int* __restrict__ sel, float* __restrict__ rweights,
                              float* __restrict__ zbuf, int* __restrict__ counts) {
  int t = blockIdx.x * blockDim.x + threadIdx.x;
  if (t >= TKN) return;
  float acc[NE];
#pragma unroll
  for (int e = 0; e < NE; ++e) acc[e] = 0.f;
  const float4* xr = (const float4*)(x + (size_t)t * H);
  const float4* wr = (const float4*)rw;
  for (int h = 0; h < H / 4; ++h) {
    float4 xv = xr[h];
#pragma unroll
    for (int e = 0; e < NE; ++e) {
      float4 wv = wr[e * (H / 4) + h];
      acc[e] += xv.x * wv.x + xv.y * wv.y + xv.z * wv.z + xv.w * wv.w;
    }
  }
  float z = 0.f, mx = acc[0];
#pragma unroll
  for (int e = 0; e < NE; ++e) { z += acc[e] * acc[e]; mx = fmaxf(mx, acc[e]); }
  zbuf[t] = z;
  float p[NE];
#pragma unroll
  for (int e = 0; e < NE; ++e) p[e] = __expf(acc[e] - mx);
  // top-2, strict '>' ascending == jax top_k tie-breaking (lower index wins)
  int i0 = 0; float b0 = p[0];
#pragma unroll
  for (int e = 1; e < NE; ++e) if (p[e] > b0) { b0 = p[e]; i0 = e; }
  int i1 = (i0 == 0) ? 1 : 0; float b1 = p[i1];
#pragma unroll
  for (int e = 0; e < NE; ++e) if (e != i0 && p[e] > b1) { b1 = p[e]; i1 = e; }
  float inv = 1.f / (b0 + b1);   // exp-ratio == prob-ratio
  sel[2 * t] = i0;  sel[2 * t + 1] = i1;
  rweights[2 * t] = b0 * inv;  rweights[2 * t + 1] = b1 * inv;
  atomicAdd(&counts[i0], 1);
  atomicAdd(&counts[i1], 1);
}

__global__ void prefix_kernel(int* counts, int* bases) {
  if (threadIdx.x == 0) {
    int s = 0;
    for (int e = 0; e < NE; ++e) { bases[e] = s; s += counts[e]; }
    bases[NE] = s;           // total (== NSLOT)
    bases[NE + 1] = 0;       // zero base for the shared-expert pass
    counts[NE] = TKN;        // "count" for the shared-expert pass
  }
}

__global__ void scatter_kernel(const int* __restrict__ sel, const int* __restrict__ bases,
                               int* __restrict__ cursor, int* __restrict__ token_list,
                               int* __restrict__ id_list, int* __restrict__ slot_of) {
  int t = blockIdx.x * blockDim.x + threadIdx.x;
  if (t >= TKN) return;
  id_list[t] = t;
#pragma unroll
  for (int k = 0; k < TOPK; ++k) {
    int e = sel[2 * t + k];
    int pos = atomicAdd(&cursor[e], 1);
    int s = bases[e] + pos;
    token_list[s] = t;
    slot_of[2 * t + k] = s;
  }
}

__global__ void convert_x(const float* __restrict__ x, unsigned short* __restrict__ xbf) {
  size_t o = ((size_t)blockIdx.x * blockDim.x + threadIdx.x) * 4;
  if (o >= (size_t)TKN * H) return;
  float4 v = *(const float4*)(x + o);
  uint2 r; r.x = cvt_pk_bf16(v.x, v.y); r.y = cvt_pk_bf16(v.z, v.w);
  *(uint2*)(xbf + o) = r;
}

// ---------------- fused gate/up GEMM + SwiGLU (grouped, gathered rows) ----------------
// Double-buffered LDS (padded stride), async A-tile staging, one barrier per K step.
__global__ __launch_bounds__(256)
void gateup_kernel(const unsigned short* __restrict__ xbf,
                   const float* __restrict__ Wg, const float* __restrict__ Wu,
                   const int* __restrict__ token_list,
                   const int* __restrict__ counts, const int* __restrict__ bases,
                   unsigned short* __restrict__ act) {
  const int e   = blockIdx.z;
  const int cnt = counts[e];
  const int m0  = blockIdx.y * BM;
  if (m0 >= cnt) return;
  const int base = bases[e];
  const int n0   = blockIdx.x * BN;
  const float* wg = Wg + (size_t)e * IDIM * H;
  const float* wu = Wu + (size_t)e * IDIM * H;

  __shared__ __align__(16) unsigned short sA [2][BM * PK];
  __shared__ __align__(16) unsigned short sBg[2][BN * PK];
  __shared__ __align__(16) unsigned short sBu[2][BN * PK];

  const int tid = threadIdx.x;
  // A loader: 2 threads per row, 16 bf16 (32B) each, via 2 async b128 ops
  const int ar = tid >> 1, ah = tid & 1;
  int arow = m0 + ar; if (arow >= cnt) arow = cnt - 1;
  const int tok = token_list[base + arow];
  const unsigned short* asrc = xbf + (size_t)tok * H + ah * 16;
  const int aoff = ar * PK + ah * 16;
  // B loader: 4 threads per row, 8 f32 -> 8 bf16 each
  const int br = tid >> 2, bq = tid & 3;
  const float* bgsrc = wg + (size_t)(n0 + br) * H + bq * 8;
  const float* busrc = wu + (size_t)(n0 + br) * H + bq * 8;
  const int boff = br * PK + bq * 8;

  // wave tiling: 8 waves -> 4 (M) x 2 (N) of 32x32
  const int lane = tid & 31, wave = tid >> 5;
  const int mbase = (wave & 3) * 32, nbase = (wave >> 2) * 32;
  const int ln = lane & 15, hi = lane >> 4;
  const int aF = (mbase + ln) * PK + hi * 16;   // + mi*16*PK ; +8 for elems 8..15
  const int bF = (nbase + ln) * PK + hi * 16;   // + ni*16*PK

  const floatx8 vz = {0.f,0.f,0.f,0.f,0.f,0.f,0.f,0.f};
  floatx8 accg[2][2], accu[2][2];
#pragma unroll
  for (int a = 0; a < 2; ++a)
#pragma unroll
    for (int b = 0; b < 2; ++b) { accg[a][b] = vz; accu[a][b] = vz; }

  // prologue: stage K step 0 into buffer 0
  float4 pg0, pg1, pu0, pu1;
  async_copy_b128(asrc,     &sA[0][aoff]);
  async_copy_b128(asrc + 8, &sA[0][aoff + 8]);
  pg0 = ((const float4*)bgsrc)[0]; pg1 = ((const float4*)bgsrc)[1];
  pu0 = ((const float4*)busrc)[0]; pu1 = ((const float4*)busrc)[1];
  *(uint4*)&sBg[0][boff] = pack8(pg0, pg1);
  *(uint4*)&sBu[0][boff] = pack8(pu0, pu1);
  wait_async0();

  const int NK = H / BK;
  for (int k = 0; k < NK; ++k) {
    __syncthreads();
    const int cur = k & 1, nxt = cur ^ 1;
    const bool more = (k + 1) < NK;
    if (more) {  // prefetch next tile; safe: everyone passed barrier k
      const int kk = (k + 1) * BK;
      async_copy_b128(asrc + kk,     &sA[nxt][aoff]);
      async_copy_b128(asrc + kk + 8, &sA[nxt][aoff + 8]);
      pg0 = ((const float4*)(bgsrc + kk))[0]; pg1 = ((const float4*)(bgsrc + kk))[1];
      pu0 = ((const float4*)(busrc + kk))[0]; pu1 = ((const float4*)(busrc + kk))[1];
    }
    Frag fa[2], fg[2], fu[2];
#pragma unroll
    for (int mi = 0; mi < 2; ++mi) {
      fa[mi].q[0] = *(const uint4*)&sA[cur][aF + mi * 16 * PK];
      fa[mi].q[1] = *(const uint4*)&sA[cur][aF + mi * 16 * PK + 8];
    }
#pragma unroll
    for (int ni = 0; ni < 2; ++ni) {
      const uint4* pg = (const uint4*)&sBg[cur][bF + ni * 16 * PK];
      fg[ni].q[0] = pg[0]; fg[ni].q[1] = pg[1];
      const uint4* pu = (const uint4*)&sBu[cur][bF + ni * 16 * PK];
      fu[ni].q[0] = pu[0]; fu[ni].q[1] = pu[1];
    }
#pragma unroll
    for (int mi = 0; mi < 2; ++mi)
#pragma unroll
      for (int ni = 0; ni < 2; ++ni) {
        accg[mi][ni] = __builtin_amdgcn_wmma_f32_16x16x32_bf16(
            false, fa[mi].v, false, fg[ni].v, (short)0, accg[mi][ni], false, false);
        accu[mi][ni] = __builtin_amdgcn_wmma_f32_16x16x32_bf16(
            false, fa[mi].v, false, fu[ni].v, (short)0, accu[mi][ni], false, false);
      }
    if (more) {
      *(uint4*)&sBg[nxt][boff] = pack8(pg0, pg1);
      *(uint4*)&sBu[nxt][boff] = pack8(pu0, pu1);
      wait_async0();
    }
  }

  // epilogue: act = silu(g) * u  (C layout: lane<16 -> M=j, lane>=16 -> M=j+8)
#pragma unroll
  for (int mi = 0; mi < 2; ++mi)
#pragma unroll
    for (int ni = 0; ni < 2; ++ni) {
      int col = n0 + nbase + ni * 16 + ln;
#pragma unroll
      for (int j = 0; j < 8; ++j) {
        int gm = m0 + mbase + mi * 16 + j + hi * 8;
        if (gm < cnt) {
          float g = accg[mi][ni][j], u = accu[mi][ni][j];
          float s = g * __builtin_amdgcn_rcpf(1.f + __expf(-g));
          act[(size_t)(base + gm) * IDIM + col] = f2bf(s * u);
        }
      }
    }
}

// ---------------- down GEMM (grouped, contiguous slot rows) ----------------
__global__ __launch_bounds__(256)
void down_kernel(const unsigned short* __restrict__ actv,
                 const float* __restrict__ Wd,
                 const int* __restrict__ counts, const int* __restrict__ bases,
                 float* __restrict__ out) {
  const int e   = blockIdx.z;
  const int cnt = counts[e];
  const int m0  = blockIdx.y * BM;
  if (m0 >= cnt) return;
  const int base = bases[e];
  const int n0   = blockIdx.x * BN;
  const float* wd = Wd + (size_t)e * H * IDIM;

  __shared__ __align__(16) unsigned short sA[2][BM * PK];
  __shared__ __align__(16) unsigned short sB[2][BN * PK];

  const int tid = threadIdx.x;
  const int ar = tid >> 1, ah = tid & 1;
  int arow = m0 + ar; if (arow >= cnt) arow = cnt - 1;
  const unsigned short* asrc = actv + (size_t)(base + arow) * IDIM + ah * 16;
  const int aoff = ar * PK + ah * 16;
  const int br = tid >> 2, bq = tid & 3;
  const float* bsrc = wd + (size_t)(n0 + br) * IDIM + bq * 8;
  const int boff = br * PK + bq * 8;

  const int lane = tid & 31, wave = tid >> 5;
  const int mbase = (wave & 3) * 32, nbase = (wave >> 2) * 32;
  const int ln = lane & 15, hi = lane >> 4;
  const int aF = (mbase + ln) * PK + hi * 16;
  const int bF = (nbase + ln) * PK + hi * 16;

  const floatx8 vz = {0.f,0.f,0.f,0.f,0.f,0.f,0.f,0.f};
  floatx8 acc[2][2];
#pragma unroll
  for (int a = 0; a < 2; ++a)
#pragma unroll
    for (int b = 0; b < 2; ++b) acc[a][b] = vz;

  float4 pb0, pb1;
  async_copy_b128(asrc,     &sA[0][aoff]);
  async_copy_b128(asrc + 8, &sA[0][aoff + 8]);
  pb0 = ((const float4*)bsrc)[0]; pb1 = ((const float4*)bsrc)[1];
  *(uint4*)&sB[0][boff] = pack8(pb0, pb1);
  wait_async0();

  const int NK = IDIM / BK;
  for (int k = 0; k < NK; ++k) {
    __syncthreads();
    const int cur = k & 1, nxt = cur ^ 1;
    const bool more = (k + 1) < NK;
    if (more) {
      const int kk = (k + 1) * BK;
      async_copy_b128(asrc + kk,     &sA[nxt][aoff]);
      async_copy_b128(asrc + kk + 8, &sA[nxt][aoff + 8]);
      pb0 = ((const float4*)(bsrc + kk))[0]; pb1 = ((const float4*)(bsrc + kk))[1];
    }
    Frag fa[2], fb[2];
#pragma unroll
    for (int mi = 0; mi < 2; ++mi) {
      fa[mi].q[0] = *(const uint4*)&sA[cur][aF + mi * 16 * PK];
      fa[mi].q[1] = *(const uint4*)&sA[cur][aF + mi * 16 * PK + 8];
    }
#pragma unroll
    for (int ni = 0; ni < 2; ++ni) {
      const uint4* pb = (const uint4*)&sB[cur][bF + ni * 16 * PK];
      fb[ni].q[0] = pb[0]; fb[ni].q[1] = pb[1];
    }
#pragma unroll
    for (int mi = 0; mi < 2; ++mi)
#pragma unroll
      for (int ni = 0; ni < 2; ++ni)
        acc[mi][ni] = __builtin_amdgcn_wmma_f32_16x16x32_bf16(
            false, fa[mi].v, false, fb[ni].v, (short)0, acc[mi][ni], false, false);
    if (more) {
      *(uint4*)&sB[nxt][boff] = pack8(pb0, pb1);
      wait_async0();
    }
  }

#pragma unroll
  for (int mi = 0; mi < 2; ++mi)
#pragma unroll
    for (int ni = 0; ni < 2; ++ni) {
      int col = n0 + nbase + ni * 16 + ln;
#pragma unroll
      for (int j = 0; j < 8; ++j) {
        int gm = m0 + mbase + mi * 16 + j + hi * 8;
        if (gm < cnt) out[(size_t)(base + gm) * H + col] = acc[mi][ni][j];
      }
    }
}

// ---------------- combine + aux ----------------
__global__ void combine_kernel(const float* __restrict__ yr, const int* __restrict__ slot_of,
                               const float* __restrict__ rw, float* __restrict__ out) {
  int gid = blockIdx.x * blockDim.x + threadIdx.x;  // one thread per 4 elements
  int t = gid >> 8;                 // H/4 == 256 quads per token
  int hq = (gid & 255) * 4;
  if (t >= TKN) return;
  int s0 = slot_of[2 * t], s1 = slot_of[2 * t + 1];
  float w0 = rw[2 * t], w1 = rw[2 * t + 1];
  float4 a = *(const float4*)(yr + (size_t)s0 * H + hq);
  float4 b = *(const float4*)(yr + (size_t)s1 * H + hq);
  float4 o = *(const float4*)(out + (size_t)t * H + hq);   // shared-expert output
  o.x += w0 * a.x + w1 * b.x;
  o.y += w0 * a.y + w1 * b.y;
  o.z += w0 * a.z + w1 * b.z;
  o.w += w0 * a.w + w1 * b.w;
  *(float4*)(out + (size_t)t * H + hq) = o;
}

__global__ void aux_kernel(const float* __restrict__ zbuf, const int* __restrict__ counts,
                           float* __restrict__ out_scalar) {
  __shared__ float red[256];
  int tid = threadIdx.x;
  float s = 0.f;
  for (int i = 0; i < TKN / 256; ++i) s += zbuf[tid * (TKN / 256) + i];
  red[tid] = s;
  __syncthreads();
  for (int w = 128; w > 0; w >>= 1) {
    if (tid < w) red[tid] += red[tid + w];
    __syncthreads();
  }
  if (tid == 0) {
    float zsum = red[0];
    float loads[NE + 1], total = 0.f;
    for (int e = 0; e < NE; ++e) { loads[e] = (float)counts[e]; total += loads[e]; }
    loads[NE] = (float)TKN; total += loads[NE];
    float ideal = 1.f / (NE + 1), lb = 0.f;
    for (int e = 0; e <= NE; ++e) { float d = loads[e] / total - ideal; lb += d * d; }
    lb /= (float)(NE + 1);
    float zl = zsum / (float)TKN;
    *out_scalar = 0.01f * lb + 0.01f * zl;
  }
}

// ---------------- launch ----------------
extern "C" void kernel_launch(void* const* d_in, const int* in_sizes, int n_in,
                              void* d_out, int out_size, void* d_ws, size_t ws_size,
                              hipStream_t stream) {
  const float* x        = (const float*)d_in[0];
  const float* router_w = (const float*)d_in[1];
  const float* gate_w   = (const float*)d_in[2];
  const float* up_w     = (const float*)d_in[3];
  const float* down_w   = (const float*)d_in[4];
  const float* shg      = (const float*)d_in[5];
  const float* shu      = (const float*)d_in[6];
  const float* shd      = (const float*)d_in[7];
  float* out = (float*)d_out;

  // workspace carve (~137 MB total)
  char* p = (char*)d_ws;
  auto carve = [&](size_t bytes) -> char* {
    char* r = p; p += (bytes + 255) & ~(size_t)255; return r;
  };
  int*   counts  = (int*)carve(16 * sizeof(int));
  int*   bases   = (int*)carve(16 * sizeof(int));
  int*   cursor  = (int*)carve(16 * sizeof(int));
  int*   sel     = (int*)carve((size_t)NSLOT * sizeof(int));
  float* rwts    = (float*)carve((size_t)NSLOT * sizeof(float));
  float* zbuf    = (float*)carve((size_t)TKN * sizeof(float));
  int*   tlist   = (int*)carve((size_t)NSLOT * sizeof(int));
  int*   idlist  = (int*)carve((size_t)TKN * sizeof(int));
  int*   slot_of = (int*)carve((size_t)NSLOT * sizeof(int));
  unsigned short* xbf  = (unsigned short*)carve((size_t)TKN * H * 2);       //  8 MB
  unsigned short* actr = (unsigned short*)carve((size_t)NSLOT * IDIM * 2);  // 64 MB
  unsigned short* acts = (unsigned short*)carve((size_t)TKN * IDIM * 2);    // 32 MB
  float*          yr   = (float*)carve((size_t)NSLOT * H * sizeof(float));  // 32 MB

  init_meta<<<1, 64, 0, stream>>>(counts, cursor);
  router_kernel<<<TKN / 128, 128, 0, stream>>>(x, router_w, sel, rwts, zbuf, counts);
  prefix_kernel<<<1, 1, 0, stream>>>(counts, bases);
  scatter_kernel<<<TKN / 256, 256, 0, stream>>>(sel, bases, cursor, tlist, idlist, slot_of);
  convert_x<<<(TKN * H / 4) / 256, 256, 0, stream>>>(x, xbf);

  // routed experts
  dim3 ggr(IDIM / BN, TKN / BM, NE);
  gateup_kernel<<<ggr, 256, 0, stream>>>(xbf, gate_w, up_w, tlist, counts, bases, actr);
  dim3 gdr(H / BN, TKN / BM, NE);
  down_kernel<<<gdr, 256, 0, stream>>>(actr, down_w, counts, bases, yr);

  // shared expert (identity token list, count = T, base = 0, writes d_out)
  dim3 ggs(IDIM / BN, TKN / BM, 1);
  gateup_kernel<<<ggs, 256, 0, stream>>>(xbf, shg, shu, idlist, counts + NE, bases + NE + 1, acts);
  dim3 gds(H / BN, TKN / BM, 1);
  down_kernel<<<gds, 256, 0, stream>>>(acts, shd, counts + NE, bases + NE + 1, out);

  combine_kernel<<<(TKN * H / 4) / 256, 256, 0, stream>>>(yr, slot_of, rwts, out);
  aux_kernel<<<1, 256, 0, stream>>>(zbuf, counts, out + (size_t)TKN * H);
}
